// Joiner_28372553957864
// MI455X (gfx1250) — matrix operand order repaired and checked by reading, not compiled, not run on hardware
//
#include <hip/hip_runtime.h>
#include <hip/hip_bf16.h>
#include <math.h>

typedef __attribute__((ext_vector_type(16))) __bf16 v16bf;
typedef __attribute__((ext_vector_type(8)))  __bf16 v8bf;
typedef __attribute__((ext_vector_type(8)))  float  v8f;

#define LDS_STRIDE   520      // 512 + 8 bf16 pad -> row stride 260 dwords, spreads LDS banks
#define ROWS_PER_BLK 64
#define K_STEPS      16       // 512 / 32
#define V_DIM        500

// ---------------------------------------------------------------------------
// Pack a (vrows x 512) row-major f32 weight matrix into bf16 WMMA B-fragment
// order. Fragment layout (v_wmma_*_16x16x32, wave32, 32x16 B tile):
//   lane 0-15  : column N = lane,     elements i=0..15 -> K = kstep*32 + i
//   lane 16-31 : column N = lane-16,  elements i=0..15 -> K = kstep*32 + 16 + i
// Packed address = ((kstep*32 + ntile)*32 + lane)*16 elements -> each lane
// reads 32 contiguous bytes; a full fragment read is 1KB contiguous.
// ---------------------------------------------------------------------------
__global__ __launch_bounds__(256) void pack_weight_kernel(
    const float* __restrict__ src, __bf16* __restrict__ dst, int vrows) {
  int tid = blockIdx.x * 256 + threadIdx.x;
  if (tid >= 512 * 512) return;
  int p     = tid >> 9;        // kstep*32 + ntile
  int e     = tid & 511;
  int lane  = e >> 4;
  int i     = e & 15;
  int kstep = p >> 5;
  int nt    = p & 31;
  int v = nt * 16 + (lane & 15);
  int j = kstep * 32 + ((lane >= 16) ? 16 : 0) + i;
  float val = (v < vrows) ? src[v * 512 + j] : 0.0f;
  dst[tid] = (__bf16)val;
}

// ---------------------------------------------------------------------------
// Core WMMA loop: C(64x64 per wave-group view) += A(64x512 in LDS) * B.
// A fragment layout used here (16-bit A, 16x32, contiguous K, same as B packing):
//   lane row M = lane%16; lanes 0-15: K = k*32+0..15
//                         lanes 16-31: K = k*32+16..31
// acc[m][j] : M-tile m (0..3), local N-tile j (0..3) of this wave.
// ---------------------------------------------------------------------------
__device__ __forceinline__ void gemm_tiles(const __bf16* sA,
                                           const __bf16* __restrict__ Wp,
                                           int lane, int ntBase, v8f acc[4][4]) {
  const int hi = (lane >= 16) ? 1 : 0;
  const int lr = lane & 15;
  for (int k = 0; k < K_STEPS; ++k) {
    v16bf a[4];
#pragma unroll
    for (int m = 0; m < 4; ++m) {
      const v8bf* pa =
          (const v8bf*)(sA + (m * 16 + lr) * LDS_STRIDE + k * 32 + hi * 16);
      v8bf lo = pa[0];          // K base .. base+7      (ds_load_b128)
      v8bf hi8 = pa[1];         // K base+8 .. base+15   (ds_load_b128)
      a[m] = __builtin_shufflevector(lo, hi8, 0, 1, 2, 3, 4, 5, 6, 7, 8, 9, 10,
                                     11, 12, 13, 14, 15);
    }
#pragma unroll
    for (int j = 0; j < 4; ++j) {
      const v16bf b =
          *(const v16bf*)(Wp + (((k * 32) + (ntBase + j)) * 32 + lane) * 16);
#pragma unroll
      for (int m = 0; m < 4; ++m)
        acc[m][j] = __builtin_amdgcn_wmma_f32_16x16x32_bf16(
            false, a[m], false, b, (short)0, acc[m][j], false, false);
    }
  }
}

// ---------------------------------------------------------------------------
// Projection GEMM: P(nrows x 512) = X(nrows x 512) @ Wp^T   (biases deferred)
// ---------------------------------------------------------------------------
__global__ __launch_bounds__(256) void proj_gemm_kernel(
    const float* __restrict__ X, const __bf16* __restrict__ Wp,
    float* __restrict__ P, int nrows) {
  __shared__ __bf16 sA[ROWS_PER_BLK * LDS_STRIDE];
  const int row0 = blockIdx.x * ROWS_PER_BLK;
  const int tid = threadIdx.x;

  for (int i = tid; i < ROWS_PER_BLK * 128; i += 256) {
    int r = i >> 7, c4 = i & 127;
    int rg = row0 + r;
    if (rg >= nrows) rg = nrows - 1;               // clamp for load
    float4 x4 = ((const float4*)(X + rg * 512))[c4];
    __bf16* d = &sA[r * LDS_STRIDE + c4 * 4];
    d[0] = (__bf16)x4.x; d[1] = (__bf16)x4.y;
    d[2] = (__bf16)x4.z; d[3] = (__bf16)x4.w;
  }
  __syncthreads();

  const int lane = tid & 31;
  const int ntBase = (tid >> 5) * 4;               // wave owns 4 N-tiles
  v8f acc[4][4];
  const v8f zero = {0.f, 0.f, 0.f, 0.f, 0.f, 0.f, 0.f, 0.f};
#pragma unroll
  for (int m = 0; m < 4; ++m)
#pragma unroll
    for (int j = 0; j < 4; ++j) acc[m][j] = zero;

  gemm_tiles(sA, Wp, lane, ntBase, acc);

  const int hi = (lane >= 16) ? 1 : 0;
  const int lr = lane & 15;
#pragma unroll
  for (int j = 0; j < 4; ++j) {
    const int col = (ntBase + j) * 16 + lr;
#pragma unroll
    for (int m = 0; m < 4; ++m) {
#pragma unroll
      for (int r8 = 0; r8 < 8; ++r8) {
        const int rowg = row0 + m * 16 + r8 + hi * 8;
        if (rowg < nrows) P[rowg * 512 + col] = acc[m][j][r8];
      }
    }
  }
}

// ---------------------------------------------------------------------------
// Fused joiner: act = tanh(enc_p[n,t,:] + dec_p[n,u,:] + b_enc + b_dec) kept
// only in LDS (bf16); logits = act @ W_out^T + b_out streamed straight out.
// 64 rows/block -> 1250 blocks; W_out read exactly once per block from L2.
// ---------------------------------------------------------------------------
__global__ __launch_bounds__(256) void joiner_main_kernel(
    const float* __restrict__ encP, const float* __restrict__ decP,
    const float* __restrict__ bEnc, const float* __restrict__ bDec,
    const __bf16* __restrict__ WoutP, const float* __restrict__ bOut,
    float* __restrict__ out) {
  __shared__ __bf16 sA[ROWS_PER_BLK * LDS_STRIDE];
  const int row0 = blockIdx.x * ROWS_PER_BLK;
  const int tid = threadIdx.x;

  for (int i = tid; i < ROWS_PER_BLK * 128; i += 256) {
    int r = i >> 7, c4 = i & 127;
    int rg = row0 + r;                 // global (n,t,u) row, always < 80000
    int n = rg / 10000;
    int rem = rg - n * 10000;
    int t = rem / 50;
    int u = rem - t * 50;
    const float4 e4 = ((const float4*)(encP + (n * 200 + t) * 512))[c4];
    const float4 d4 = ((const float4*)(decP + (n * 50 + u) * 512))[c4];
    const float4 be = ((const float4*)bEnc)[c4];
    const float4 bd = ((const float4*)bDec)[c4];
    __bf16* d = &sA[r * LDS_STRIDE + c4 * 4];
    d[0] = (__bf16)tanhf(e4.x + d4.x + be.x + bd.x);
    d[1] = (__bf16)tanhf(e4.y + d4.y + be.y + bd.y);
    d[2] = (__bf16)tanhf(e4.z + d4.z + be.z + bd.z);
    d[3] = (__bf16)tanhf(e4.w + d4.w + be.w + bd.w);
  }
  __syncthreads();

  const int lane = tid & 31;
  const int ntBase = (tid >> 5) * 4;
  v8f acc[4][4];
  const v8f zero = {0.f, 0.f, 0.f, 0.f, 0.f, 0.f, 0.f, 0.f};
#pragma unroll
  for (int m = 0; m < 4; ++m)
#pragma unroll
    for (int j = 0; j < 4; ++j) acc[m][j] = zero;

  gemm_tiles(sA, WoutP, lane, ntBase, acc);

  const int hi = (lane >= 16) ? 1 : 0;
  const int lr = lane & 15;
#pragma unroll
  for (int j = 0; j < 4; ++j) {
    const int col = (ntBase + j) * 16 + lr;
    if (col < V_DIM) {
      const float bo = bOut[col];
#pragma unroll
      for (int m = 0; m < 4; ++m) {
#pragma unroll
        for (int r8 = 0; r8 < 8; ++r8) {
          const int rowg = row0 + m * 16 + r8 + hi * 8;
          out[rowg * V_DIM + col] = acc[m][j][r8] + bo;
        }
      }
    }
  }
}

// ---------------------------------------------------------------------------
// Workspace layout (bytes):
//   [0       , 524288 )  W_out packed bf16 (512x512)
//   [524288  , 1048576)  W_enc packed bf16
//   [1048576 , 1572864)  W_dec packed bf16
//   [1572864 , 4849664)  enc_p f32 (1600 x 512)
//   [4849664 , 5668864)  dec_p f32 (400 x 512)
// ---------------------------------------------------------------------------
extern "C" void kernel_launch(void* const* d_in, const int* in_sizes, int n_in,
                              void* d_out, int out_size, void* d_ws,
                              size_t ws_size, hipStream_t stream) {
  (void)in_sizes; (void)n_in; (void)out_size; (void)ws_size;
  const float* enc   = (const float*)d_in[0];
  const float* dec   = (const float*)d_in[1];
  const float* W_enc = (const float*)d_in[2];
  const float* b_enc = (const float*)d_in[3];
  const float* W_dec = (const float*)d_in[4];
  const float* b_dec = (const float*)d_in[5];
  const float* W_out = (const float*)d_in[6];
  const float* b_out = (const float*)d_in[7];
  float* out = (float*)d_out;

  char* ws = (char*)d_ws;
  __bf16* wOutP = (__bf16*)(ws + 0);
  __bf16* wEncP = (__bf16*)(ws + 524288);
  __bf16* wDecP = (__bf16*)(ws + 1048576);
  float*  encP  = (float*)(ws + 1572864);
  float*  decP  = (float*)(ws + 4849664);

  dim3 blk(256);
  pack_weight_kernel<<<1024, blk, 0, stream>>>(W_out, wOutP, 500);
  pack_weight_kernel<<<1024, blk, 0, stream>>>(W_enc, wEncP, 512);
  pack_weight_kernel<<<1024, blk, 0, stream>>>(W_dec, wDecP, 512);
  proj_gemm_kernel<<<25, blk, 0, stream>>>(enc, wEncP, encP, 1600);   // N*T
  proj_gemm_kernel<<<7,  blk, 0, stream>>>(dec, wDecP, decP, 400);    // N*U
  joiner_main_kernel<<<1250, blk, 0, stream>>>(encP, decP, b_enc, b_dec,
                                               wOutP, b_out, out);    // 80000/64
}
